// GroundTextBlock_context_49761491091856
// MI455X (gfx1250) — matrix operand order repaired and checked by reading, not compiled, not run on hardware
//
#include <hip/hip_runtime.h>
#include <hip/hip_bf16.h>
#include <math.h>

// Shapes from reference
#define B_    32
#define N_    20
#define D_    512
#define EMB_  512
#define HW_   4096
#define TEMP_ 10.0f

typedef __bf16 bf16_t;
typedef bf16_t  bf16x16 __attribute__((ext_vector_type(16)));
typedef float   floatx8 __attribute__((ext_vector_type(8)));

__device__ __forceinline__ unsigned short f32_to_bf16(float f) {
    unsigned int u = __float_as_uint(f);
    unsigned int r = u + 0x7FFFu + ((u >> 16) & 1u);   // round-to-nearest-even
    return (unsigned short)(r >> 16);
}
__device__ __forceinline__ unsigned int pack_bf16x2(float lo, float hi) {
    return (unsigned int)f32_to_bf16(lo) | ((unsigned int)f32_to_bf16(hi) << 16);
}

// ---------------------------------------------------------------------------
// Kernel 1: tile_visu[b,d] = mean over spatial of fvisu  (one block per (b,d))
// ---------------------------------------------------------------------------
__global__ __launch_bounds__(256) void spatial_mean_kernel(
    const float* __restrict__ x, float* __restrict__ out) {
    __shared__ float red[256];
    int bc = blockIdx.x;
    const float* p = x + (size_t)bc * HW_;
    float s = 0.f;
    for (int i = threadIdx.x; i < HW_; i += 256) s += p[i];
    red[threadIdx.x] = s;
    __syncthreads();
    for (int off = 128; off > 0; off >>= 1) {
        if (threadIdx.x < off) red[threadIdx.x] += red[threadIdx.x + off];
        __syncthreads();
    }
    if (threadIdx.x == 0) out[bc] = red[0] * (1.0f / HW_);
}

// ---------------------------------------------------------------------------
// Kernel 2: fused attention / FiLM path (one block per batch, 256 threads)
// Produces: attn_lang_my (out2), attn_score (out3), gammas/betas (ws)
// ---------------------------------------------------------------------------
__global__ __launch_bounds__(256) void attn_film_kernel(
    const float* __restrict__ fword, const float* __restrict__ context_score,
    const float* __restrict__ attn_map_w, const float* __restrict__ attn_map_b,
    const float* __restrict__ attn_score_w, const float* __restrict__ attn_score_b,
    const float* __restrict__ gamma_w, const float* __restrict__ gamma_b,
    const int* __restrict__ word_mask, const float* __restrict__ tile_visu,
    float* __restrict__ out_attn_lang_my, float* __restrict__ out_attn_score,
    float* __restrict__ gammas, float* __restrict__ betas) {

    __shared__ float fwS[N_ * D_];    // 40 KB: fword[b] as [n][d]
    __shared__ float pool[256 * N_];  // 20 KB: phase A = attn_feat[o][n]; phase B = attn_lang[d]
    __shared__ float tvS[D_];
    __shared__ float csS[N_];
    __shared__ float msS[N_];         // raw scores, then mask_score

    int b = blockIdx.x;
    int tid = threadIdx.x;

    for (int i = tid; i < N_ * D_; i += 256) fwS[i] = fword[(size_t)b * N_ * D_ + i];
    for (int i = tid; i < D_; i += 256) tvS[i] = tile_visu[b * D_ + i];
    if (tid < N_) csS[tid] = context_score[b * N_ + tid];
    __syncthreads();

    // attn_feat[o,n] = tanh( sum_d W[o,d]*tv[d]*fword[n,d] * cs[n] + b[o] )
    for (int task = tid; task < 256 * N_; task += 256) {
        int o = task / N_, n = task % N_;
        const float* wr = &attn_map_w[(size_t)o * D_];
        const float* fr = &fwS[n * D_];
        float s = 0.f;
        for (int d = 0; d < D_; ++d) s = fmaf(wr[d] * tvS[d], fr[d], s);
        pool[o * N_ + n] = tanhf(s * csS[n] + attn_map_b[o]);
    }
    __syncthreads();

    // attn_score[n] (raw -> output 3)
    if (tid < N_) {
        float s = attn_score_b[0];
        for (int o = 0; o < 256; ++o) s = fmaf(attn_score_w[o], pool[o * N_ + tid], s);
        out_attn_score[b * N_ + tid] = s;
        msS[tid] = s;
    }
    __syncthreads();

    // mask_softmax (tiny, single lane)
    if (tid == 0) {
        float sc[N_], m[N_];
        float mx = -1e30f;
        for (int n = 0; n < N_; ++n) { sc[n] = msS[n] * TEMP_; mx = fmaxf(mx, sc[n]); }
        float ssum = 0.f;
        for (int n = 0; n < N_; ++n) { sc[n] = __expf(sc[n] - mx); ssum += sc[n]; }
        float inv0 = 1.f / ssum;
        float msum = 0.f;
        for (int n = 0; n < N_; ++n) {
            sc[n] *= inv0;
            m[n] = (n == 0) ? 0.f : (float)word_mask[b * N_ + n];
            msum += m[n];
        }
        int sInt = (int)msum;                      // SEP position
        if (sInt >= 0 && sInt < N_) m[sInt] = 0.f;
        float tot = 0.f;
        for (int n = 0; n < N_; ++n) { sc[n] *= m[n]; tot += sc[n]; }
        float inv = 1.f / (tot + 1e-8f);
        for (int n = 0; n < N_; ++n) msS[n] = sc[n] * inv;
    }
    __syncthreads();

    // attn_lang_my[b,d,n]  (output 2)
    for (int task = tid; task < D_ * N_; task += 256) {
        int d = task / N_, n = task % N_;
        out_attn_lang_my[((size_t)b * D_ + d) * N_ + n] = msS[n] * fwS[n * D_ + d];
    }
    // attn_lang[d] -> pool[0..511] (pool re-used; attn_feat dead now)
    for (int d = tid; d < D_; d += 256) {
        float s = 0.f;
        for (int n = 0; n < N_; ++n) s = fmaf(msS[n], fwS[n * D_ + d], s);
        pool[d] = s;
    }
    __syncthreads();

    // FiLM: film[j] = attn_lang . gamma_w[j] + gamma_b[j]; tanh -> gammas/betas
    for (int j = tid; j < 2 * EMB_; j += 256) {
        const float* gr = &gamma_w[(size_t)j * D_];
        float s = gamma_b[j];
        for (int d = 0; d < D_; ++d) s = fmaf(gr[d], pool[d], s);
        float t = tanhf(s);
        if (j < EMB_) gammas[b * EMB_ + j] = t;
        else          betas[b * EMB_ + (j - EMB_)] = t;
    }
}

// ---------------------------------------------------------------------------
// Kernel 3: conv1 (1x1) as bf16 WMMA GEMM: fmodu[b,o,p] = sum_c W[o,c]*X[b,c,p]
// Block = 256 threads (8 waves) -> 256(o) x 64(p) tile; wave -> 64x32 sub-tile
// (4 A-frags x 2 B-frags = 8 WMMAs / K-step / wave). LDS double-buffered.
// ---------------------------------------------------------------------------
#define ASTRIDE 34
__global__ __launch_bounds__(256) void conv1_wmma_kernel(
    const float* __restrict__ fvisu, const float* __restrict__ w,
    const float* __restrict__ bias, float* __restrict__ out) {

    __shared__ unsigned short As[2][256 * ASTRIDE];  // W tile [o][k], padded
    __shared__ unsigned short Bs[2][64 * ASTRIDE];   // X tile transposed [p][k]

    int blk   = blockIdx.x;
    int b     = blk >> 7;                 // 128 tile-blocks per batch
    int rem   = blk & 127;
    int oBase = (rem >> 6) << 8;          // 0 or 256
    int pBase = (rem & 63) << 6;          // 0..4032 step 64
    const float* X = fvisu + (size_t)b * D_ * HW_;

    int tid  = threadIdx.x;
    int wave = tid >> 5;
    int lane = tid & 31;
    int wo   = wave >> 1;                 // 0..3 : 64-row group
    int wp   = wave & 1;                  // 0..1 : 32-col group
    int half = lane >> 4;                 // upper K/M group for lanes 16-31
    int nl   = lane & 15;

    floatx8 acc[4][2];
    #pragma unroll
    for (int i = 0; i < 4; ++i)
        #pragma unroll
        for (int j = 0; j < 2; ++j)
            #pragma unroll
            for (int v = 0; v < 8; ++v) acc[i][j][v] = 0.f;

    // ---- tile fill helper (k0 = K offset, buf = LDS buffer index) ----
    auto fill = [&](int k0, int buf) {
        // A: [256 x 32] fp32 -> bf16, float4 loads, packed 2xbf16 stores
        #pragma unroll
        for (int r = 0; r < 8; ++r) {
            int l4 = tid + 256 * r;            // 2048 float4 groups
            int o  = l4 >> 3;
            int c4 = (l4 & 7) << 2;
            float4 v4 = *(const float4*)&w[(size_t)(oBase + o) * D_ + (k0 + c4)];
            unsigned int* dst = (unsigned int*)&As[buf][o * ASTRIDE + c4];
            dst[0] = pack_bf16x2(v4.x, v4.y);
            dst[1] = pack_bf16x2(v4.z, v4.w);
        }
        // B: [32 x 64] fp32 -> bf16, transposed store [p][k]
        #pragma unroll
        for (int r = 0; r < 2; ++r) {
            int l4 = tid + 256 * r;            // 512 float4 groups
            int kk  = l4 >> 4;
            int pp4 = (l4 & 15) << 2;
            float4 v4 = *(const float4*)&X[(size_t)(k0 + kk) * HW_ + (pBase + pp4)];
            Bs[buf][(pp4 + 0) * ASTRIDE + kk] = f32_to_bf16(v4.x);
            Bs[buf][(pp4 + 1) * ASTRIDE + kk] = f32_to_bf16(v4.y);
            Bs[buf][(pp4 + 2) * ASTRIDE + kk] = f32_to_bf16(v4.z);
            Bs[buf][(pp4 + 3) * ASTRIDE + kk] = f32_to_bf16(v4.w);
        }
    };

    fill(0, 0);
    __syncthreads();

    for (int kt = 0; kt < D_ / 32; ++kt) {
        int cur = kt & 1;
        if (kt + 1 < D_ / 32) fill((kt + 1) * 32, cur ^ 1);   // prefetch next tile

        // Build fragments per ISA 7.12.2 16-bit layouts (wave32)
        union { bf16x16 v; unsigned int d[8]; } a[4], bb[2];
        #pragma unroll
        for (int i = 0; i < 4; ++i) {
            int m = wo * 64 + i * 16 + nl;                 // A row (M)
            const unsigned short* row = &As[cur][m * ASTRIDE];
            #pragma unroll
            for (int e2 = 0; e2 < 8; ++e2) {
                int e = e2 * 2;
                int k = (e < 8 ? e : e + 8) + half * 8;    // K pairs: {0-7,16-23}/{8-15,24-31}
                a[i].d[e2] = *(const unsigned int*)&row[k];
            }
        }
        #pragma unroll
        for (int j = 0; j < 2; ++j) {
            int n = wp * 32 + j * 16 + nl;                 // B column (N)
            const unsigned short* row = &Bs[cur][n * ASTRIDE];
            #pragma unroll
            for (int e2 = 0; e2 < 8; ++e2) {
                int k = (e2 < 4 ? e2 * 2 : e2 * 2 + 8) + half * 8;   // K pairs: {0-7,16-23}/{8-15,24-31}
                bb[j].d[e2] = *(const unsigned int*)&row[k];
            }
        }

        #pragma unroll
        for (int i = 0; i < 4; ++i)
            #pragma unroll
            for (int j = 0; j < 2; ++j)
                acc[i][j] = __builtin_amdgcn_wmma_f32_16x16x32_bf16(
                    false, a[i].v, false, bb[j].v, (short)0, acc[i][j], false, false);

        __syncthreads();   // next-tile fill complete AND this-tile reads done
    }

    // Epilogue: + bias, store raw fmodu (normalized later in place)
    #pragma unroll
    for (int i = 0; i < 4; ++i) {
        #pragma unroll
        for (int v = 0; v < 8; ++v) {
            int o = oBase + wo * 64 + i * 16 + half * 8 + v;   // D layout: M = v + 8*half
            float bi = bias[o];
            size_t rowOff = ((size_t)b * D_ + o) * HW_;
            #pragma unroll
            for (int j = 0; j < 2; ++j) {
                int p = pBase + wp * 32 + j * 16 + nl;         // N = lane&15
                out[rowOff + p] = acc[i][j][v] + bi;
            }
        }
    }
}

// ---------------------------------------------------------------------------
// Kernel 4: instance-norm stats per (b,c)
// ---------------------------------------------------------------------------
__global__ __launch_bounds__(256) void instnorm_stats_kernel(
    const float* __restrict__ x, float* __restrict__ meanOut, float* __restrict__ rstdOut) {
    __shared__ float rs[256], rq[256];
    int bc = blockIdx.x;
    const float* p = x + (size_t)bc * HW_;
    float s = 0.f, q = 0.f;
    for (int i = threadIdx.x; i < HW_; i += 256) { float v = p[i]; s += v; q = fmaf(v, v, q); }
    rs[threadIdx.x] = s; rq[threadIdx.x] = q;
    __syncthreads();
    for (int off = 128; off > 0; off >>= 1) {
        if (threadIdx.x < off) {
            rs[threadIdx.x] += rs[threadIdx.x + off];
            rq[threadIdx.x] += rq[threadIdx.x + off];
        }
        __syncthreads();
    }
    if (threadIdx.x == 0) {
        float mu  = rs[0] * (1.f / HW_);
        float var = rq[0] * (1.f / HW_) - mu * mu;
        meanOut[bc] = mu;
        rstdOut[bc] = rsqrtf(var + 1e-5f);
    }
}

// ---------------------------------------------------------------------------
// Kernel 5: in-place  relu(gamma*(x-mu)*rstd + beta), float4 vectorized
// ---------------------------------------------------------------------------
__global__ __launch_bounds__(256) void film_norm_kernel(
    float* __restrict__ fmodu, const float* __restrict__ mean, const float* __restrict__ rstd,
    const float* __restrict__ gammas, const float* __restrict__ betas) {
    size_t base = ((size_t)blockIdx.x * 256 + threadIdx.x) * 4;
    int bc = (int)(base >> 12);                 // 4096 pixels per (b,c), 4 | 4096
    float mu = mean[bc], rsv = rstd[bc], g = gammas[bc], be = betas[bc];
    float4 v = *(const float4*)&fmodu[base];
    float* e = (float*)&v;
    #pragma unroll
    for (int i = 0; i < 4; ++i) {
        float t = fmaf(g, (e[i] - mu) * rsv, be);
        e[i] = t > 0.f ? t : 0.f;
    }
    *(float4*)&fmodu[base] = v;
}

// ---------------------------------------------------------------------------
extern "C" void kernel_launch(void* const* d_in, const int* in_sizes, int n_in,
                              void* d_out, int out_size, void* d_ws, size_t ws_size,
                              hipStream_t stream) {
    (void)in_sizes; (void)n_in; (void)out_size; (void)ws_size;
    const float* fvisu         = (const float*)d_in[0];
    const float* fword         = (const float*)d_in[1];
    const float* context_score = (const float*)d_in[2];
    const float* attn_map_w    = (const float*)d_in[3];
    const float* attn_map_b    = (const float*)d_in[4];
    const float* attn_score_w  = (const float*)d_in[5];
    const float* attn_score_b  = (const float*)d_in[6];
    const float* gamma_w       = (const float*)d_in[7];
    const float* gamma_b       = (const float*)d_in[8];
    const float* conv1_w       = (const float*)d_in[9];
    const float* conv1_b       = (const float*)d_in[10];
    const int*   word_mask     = (const int*)d_in[11];

    float* out       = (float*)d_out;
    float* out_fmodu = out;                                    // [B,512,64,64]
    float* out_almy  = out + (size_t)B_ * D_ * HW_;            // [B,512,20]
    float* out_as    = out_almy + (size_t)B_ * D_ * N_;        // [B,20]

    float* ws        = (float*)d_ws;                           // 5 * 16384 floats
    float* tile_visu = ws;                                     // [B*D]
    float* gammas    = ws + 16384;                             // [B*EMB]
    float* betas     = ws + 2 * 16384;                         // [B*EMB]
    float* meanA     = ws + 3 * 16384;                         // [B*512]
    float* rstdA     = ws + 4 * 16384;                         // [B*512]

    spatial_mean_kernel<<<B_ * D_, 256, 0, stream>>>(fvisu, tile_visu);
    attn_film_kernel<<<B_, 256, 0, stream>>>(fword, context_score,
        attn_map_w, attn_map_b, attn_score_w, attn_score_b,
        gamma_w, gamma_b, word_mask, tile_visu,
        out_almy, out_as, gammas, betas);
    conv1_wmma_kernel<<<B_ * 128, 256, 0, stream>>>(fvisu, conv1_w, conv1_b, out_fmodu);
    instnorm_stats_kernel<<<B_ * D_, 256, 0, stream>>>(out_fmodu, meanA, rstdA);
    film_norm_kernel<<<(B_ * D_ * HW_) / (256 * 4), 256, 0, stream>>>(
        out_fmodu, meanA, rstdA, gammas, betas);
}
